// MatchRNN_16166256902620
// MI455X (gfx1250) — matrix operand order repaired and checked by reading, not compiled, not run on hardware
//
#include <hip/hip_runtime.h>
#include <stdint.h>
#include <stddef.h>

#define DEV static __device__ __forceinline__

typedef __attribute__((ext_vector_type(16))) __bf16 v16bf;
typedef __attribute__((ext_vector_type(8)))  float  v8f;

namespace {
constexpr int T_ = 384, Q_ = 64, B_ = 64, D_ = 256, H_ = 128;
constexpr int D2 = 2 * D_;   // 512
constexpr int H3 = 3 * H_;   // 384
}

union Frag16 {
  v16bf    v;
  uint16_t e[16];
  uint4    q[2];
};

DEV uint16_t f2bf(float f) {                       // f32 -> bf16, RNE
  uint32_t u = __float_as_uint(f);
  u += 0x7fffu + ((u >> 16) & 1u);
  return (uint16_t)(u >> 16);
}
DEV float bflo(uint32_t u) { return __uint_as_float(u << 16); }
DEV float bfhi(uint32_t u) { return __uint_as_float(u & 0xffff0000u); }
DEV float sigmoidf_(float x) { return 1.0f / (1.0f + expf(-x)); }

DEV v8f wmma_bf16(v16bf a, v16bf b, v8f c) {
  // D = A(16x32 bf16) * B(32x16 bf16) + C(16x16 f32)
  return __builtin_amdgcn_wmma_f32_16x16x32_bf16(false, a, false, b, (short)0, c,
                                                 false, false);
}

// ---- Fragment loaders (ISA 7.12.2 layouts, wave32) ------------------------
// A 16x32 bf16: lane L: m = m0 + L%16; VGPR j holds K = (j/4)*16+(L/16)*8+(j%4)*2
// -> two contiguous 16B runs per lane.
DEV v16bf load_a_bf16(const uint16_t* src, int lda, int m0, int kk) {
  const int lane = threadIdx.x & 31;
  const uint16_t* row = src + (size_t)(m0 + (lane & 15)) * lda + kk + ((lane >> 4) << 3);
  Frag16 a;
  a.q[0] = *(const uint4*)(row);
  a.q[1] = *(const uint4*)(row + 16);
  return a.v;
}
// B 32x16 bf16 from weight stored (N,K) row-major: lane L: n = n0 + L%16;
// element e <-> K = kk + (e/8)*16 + (L/16)*8 + e%8  (two contiguous 16B runs per lane).
DEV v16bf load_b_bf16(const uint16_t* w, int ldk, int n0, int kk) {
  const int lane = threadIdx.x & 31;
  const uint16_t* row = w + (size_t)(n0 + (lane & 15)) * ldk + kk + ((lane >> 4) << 3);
  Frag16 b;
  b.q[0] = *(const uint4*)(row);
  b.q[1] = *(const uint4*)(row + 16);
  return b.v;
}

// ---- Kernel A1: elementwise f32 -> bf16 (each element converted ONCE) -----
__global__ void k_cvt_bf16(const float* __restrict__ src, uint16_t* __restrict__ dst, int n) {
  int i = blockIdx.x * blockDim.x + threadIdx.x;
  if (i < n) dst[i] = f2bf(src[i]);
}

// ---- Kernel A2: C[M,N] = A[M,K](bf16) * W[N,K](bf16)^T + bias(f32) --------
// One wave per 16x16 tile; fragment loads are pure b128 loads, no conversion.
__global__ void k_gemm_bf16(const uint16_t* __restrict__ A, const uint16_t* __restrict__ W,
                            const float* __restrict__ bias, float* __restrict__ C,
                            int M, int N, int K) {
  int wave = (blockIdx.x * blockDim.x + threadIdx.x) >> 5;
  int tn = N >> 4;
  int m0 = (wave / tn) << 4;
  int n0 = (wave % tn) << 4;
  if (m0 >= M) return;
  v8f acc = {};
  for (int kk = 0; kk < K; kk += 32)
    acc = wmma_bf16(load_a_bf16(A, K, m0, kk), load_b_bf16(W, K, n0, kk), acc);
  int lane = threadIdx.x & 31;
  int n = n0 + (lane & 15);
  int mb = m0 + ((lane >> 4) << 3);
  float bv = bias[n];
#pragma unroll
  for (int r = 0; r < 8; ++r) C[(size_t)(mb + r) * N + n] = acc[r] + bv;
}

// ---- Kernel B: persistent recurrent scan (one workgroup per direction) ----
struct StepArgs {
  const float* cv; const int* cmask; const int* qmask; const uint16_t* qv_bf;
  float* out;
  const float*    wh[2];   // (Q,B,H) precomputed qv@wq^T+b
  const float*    wp[2];   // (T,B,H) precomputed cv@wp^T+b (unflipped)
  const uint16_t* wrw[2];  // (H,H)   bf16
  const uint16_t* glw[2];  // (2D,2D) bf16
  const uint16_t* wih[2];  // (3H,2D) bf16
  const uint16_t* whh[2];  // (3H,H)  bf16
  const float* wr_b[2]; const float* wg_w[2]; const float* wg_b[2];
  const float* gl_b[2]; const float* ln_g[2]; const float* ln_b[2];
  const float* b_ih[2]; const float* b_hh[2];
  float* h[2]; float* wrh[2]; float* z[2]; float* zg[2]; float* gi[2]; float* gh[2];
  uint16_t* zbf[2];        // bf16 shadow of the current GEMM A-matrix (B,2D)
};

DEV void wg_sync() { __threadfence(); __syncthreads(); }

__global__ void __launch_bounds__(1024) k_matchrnn(StepArgs Ar) {
  const int dir  = blockIdx.x;
  const int tid  = threadIdx.x;
  const int lane = tid & 31;
  const int wv   = tid >> 5;

  __shared__ __align__(16) uint16_t s_hbf[B_ * H_];  // h in bf16 (A-matrix for WMMA)
  __shared__ float s_alpha[B_][Q_];
  __shared__ int   s_len[B_];

  const float*    wh   = Ar.wh[dir];
  const float*    wp   = Ar.wp[dir];
  const uint16_t* wrw  = Ar.wrw[dir];
  const uint16_t* glw  = Ar.glw[dir];
  const uint16_t* wih  = Ar.wih[dir];
  const uint16_t* whh  = Ar.whh[dir];
  const float* wr_b = Ar.wr_b[dir]; const float* wg_w = Ar.wg_w[dir];
  const float* wg_b = Ar.wg_b[dir]; const float* gl_b = Ar.gl_b[dir];
  const float* ln_g = Ar.ln_g[dir]; const float* ln_b = Ar.ln_b[dir];
  const float* b_ih = Ar.b_ih[dir]; const float* b_hh = Ar.b_hh[dir];
  float* hbuf  = Ar.h[dir];   float* wrhb = Ar.wrh[dir];
  float* zbuf  = Ar.z[dir];   float* zgb  = Ar.zg[dir];
  float* gib   = Ar.gi[dir];  float* ghb  = Ar.gh[dir];
  uint16_t* zbf = Ar.zbf[dir];

  // init: sequence lengths + h0 = 0
  if (tid < B_) {
    int c = 0;
    for (int t = 0; t < T_; ++t) c += Ar.cmask[tid * T_ + t];
    s_len[tid] = c;
  }
  for (int i = tid; i < B_ * H_; i += 1024) { hbuf[i] = 0.0f; s_hbf[i] = 0; }
  wg_sync();

#pragma unroll 1
  for (int s = 0; s < T_; ++s) {
    // ---- S1: wrh = h@wr_w^T + wr_b + wp_t   (M=64,N=128,K=128; 32 tiles) ---
    {
      int m0 = (wv >> 3) << 4, n0 = (wv & 7) << 4;
      v8f acc = {};
#pragma unroll
      for (int kk = 0; kk < H_; kk += 32)
        acc = wmma_bf16(load_a_bf16(s_hbf, H_, m0, kk), load_b_bf16(wrw, H_, n0, kk), acc);
      int n = n0 + (lane & 15);
      int mb = m0 + ((lane >> 4) << 3);
      float wb = wr_b[n];
#pragma unroll
      for (int r = 0; r < 8; ++r) {
        int b = mb + r;
        int L = s_len[b];
        int t = (dir == 0) ? s : ((s < L) ? (L - 1 - s) : s);
        wrhb[b * H_ + n] = acc[r] + wb + wp[((size_t)t * B_ + b) * H_ + n];
      }
    }
    wg_sync();
    // ---- S2: alpha_raw[b][q] = sum_h wg[h]*tanh(wh+wrh) -------------------
    {
      int b  = tid & 63;
      int qb = tid >> 6;                      // 0..15; covers q = qb, qb+16, +32, +48
      const float* whb = wh + b * H_;
      const float* wrb = wrhb + b * H_;
      float a0 = 0, a1 = 0, a2 = 0, a3 = 0;
      for (int h = 0; h < H_; ++h) {
        float w = wrb[h];
        float g = wg_w[h];
        a0 += g * tanhf(whb[(qb      ) * B_ * H_ + h] + w);
        a1 += g * tanhf(whb[(qb + 16 ) * B_ * H_ + h] + w);
        a2 += g * tanhf(whb[(qb + 32 ) * B_ * H_ + h] + w);
        a3 += g * tanhf(whb[(qb + 48 ) * B_ * H_ + h] + w);
      }
      float gb = wg_b[0];
      s_alpha[b][qb     ] = Ar.qmask[b * Q_ + qb     ] ? (a0 + gb) : -1e30f;
      s_alpha[b][qb + 16] = Ar.qmask[b * Q_ + qb + 16] ? (a1 + gb) : -1e30f;
      s_alpha[b][qb + 32] = Ar.qmask[b * Q_ + qb + 32] ? (a2 + gb) : -1e30f;
      s_alpha[b][qb + 48] = Ar.qmask[b * Q_ + qb + 48] ? (a3 + gb) : -1e30f;
    }
    __syncthreads();
    // ---- S2b: softmax over q per row --------------------------------------
    if (tid < B_) {
      float mx = -1e30f;
      for (int q = 0; q < Q_; ++q) mx = fmaxf(mx, s_alpha[tid][q]);
      float sum = 0.0f;
      for (int q = 0; q < Q_; ++q) { float e = expf(s_alpha[tid][q] - mx); s_alpha[tid][q] = e; sum += e; }
      float inv = 1.0f / sum;
      for (int q = 0; q < Q_; ++q) s_alpha[tid][q] *= inv;
    }
    __syncthreads();
    // ---- S3: z = [c_t | h_alpha] (f32 + bf16 shadow) ----------------------
    for (int i = tid; i < B_ * D_; i += 1024) {             // c_t copy
      int b = i >> 8, d = i & 255;
      int L = s_len[b];
      int t = (dir == 0) ? s : ((s < L) ? (L - 1 - s) : s);
      float v = Ar.cv[((size_t)t * B_ + b) * D_ + d];
      zbuf[b * D2 + d] = v;
      zbf[b * D2 + d]  = f2bf(v);
    }
    {                                                       // h_alpha (batched per-b matvec)
      int b0 = wv << 1;
#pragma unroll
      for (int bi = 0; bi < 2; ++bi) {
        int b = b0 + bi;
        float acc[8] = {};
        for (int q = 0; q < Q_; ++q) {
          float av = s_alpha[b][q];
          const uint32_t* row = (const uint32_t*)(Ar.qv_bf + ((size_t)q * B_ + b) * D_);
#pragma unroll
          for (int j = 0; j < 4; ++j) {
            uint32_t u = row[lane + 32 * j];                // two bf16 = d, d+1
            acc[2 * j]     += av * bflo(u);
            acc[2 * j + 1] += av * bfhi(u);
          }
        }
#pragma unroll
        for (int j = 0; j < 4; ++j) {
          int d = 2 * (lane + 32 * j);
          zbuf[b * D2 + D_ + d]     = acc[2 * j];
          zbuf[b * D2 + D_ + d + 1] = acc[2 * j + 1];
          zbf[b * D2 + D_ + d]      = f2bf(acc[2 * j]);
          zbf[b * D2 + D_ + d + 1]  = f2bf(acc[2 * j + 1]);
        }
      }
    }
    wg_sync();
    // ---- S4: zg = sigmoid(z@gl_w^T + gl_b) * z  (M=64,N=512,K=512) --------
    for (int t4 = wv; t4 < 128; t4 += 32) {
      int m0 = (t4 >> 5) << 4, n0 = (t4 & 31) << 4;
      v8f acc = {};
      for (int kk = 0; kk < D2; kk += 32) {
        __builtin_prefetch(glw + (size_t)(n0 + (lane & 15)) * D2 + kk + 64, 0, 1);
        acc = wmma_bf16(load_a_bf16(zbf, D2, m0, kk), load_b_bf16(glw, D2, n0, kk), acc);
      }
      int n = n0 + (lane & 15);
      int mb = m0 + ((lane >> 4) << 3);
      float bv = gl_b[n];
#pragma unroll
      for (int r = 0; r < 8; ++r) {
        int m = mb + r;
        zgb[m * D2 + n] = sigmoidf_(acc[r] + bv) * zbuf[m * D2 + n];
      }
    }
    wg_sync();
    // ---- S5: LayerNorm rows of zg -> zbf (bf16, feeds S6 GEMMs) -----------
#pragma unroll
    for (int bi = 0; bi < 2; ++bi) {
      int b = wv * 2 + bi;
      float s1 = 0, s2 = 0;
      for (int i = lane; i < D2; i += 32) { float v = zgb[b * D2 + i]; s1 += v; s2 += v * v; }
      for (int o = 16; o; o >>= 1) { s1 += __shfl_xor(s1, o, 32); s2 += __shfl_xor(s2, o, 32); }
      float mu = s1 * (1.0f / 512.0f);
      float var = s2 * (1.0f / 512.0f) - mu * mu;
      float rstd = rsqrtf(var + 1e-5f);
      for (int i = lane; i < D2; i += 32) {
        float v = (zgb[b * D2 + i] - mu) * rstd * ln_g[i] + ln_b[i];
        zbf[b * D2 + i] = f2bf(v);
      }
    }
    wg_sync();
    // ---- S6: gi = zln@w_ih^T + b_ih ; gh = h@w_hh^T + b_hh  (M=64,N=384) --
    for (int t6 = wv; t6 < 96; t6 += 32) {
      int m0 = (t6 / 24) << 4, n0 = (t6 % 24) << 4;
      v8f accI = {};
      for (int kk = 0; kk < D2; kk += 32) {
        __builtin_prefetch(wih + (size_t)(n0 + (lane & 15)) * D2 + kk + 64, 0, 1);
        accI = wmma_bf16(load_a_bf16(zbf, D2, m0, kk), load_b_bf16(wih, D2, n0, kk), accI);
      }
      v8f accH = {};
#pragma unroll
      for (int kk = 0; kk < H_; kk += 32)
        accH = wmma_bf16(load_a_bf16(s_hbf, H_, m0, kk), load_b_bf16(whh, H_, n0, kk), accH);
      int n = n0 + (lane & 15);
      int mb = m0 + ((lane >> 4) << 3);
      float bi_ = b_ih[n], bh_ = b_hh[n];
#pragma unroll
      for (int r = 0; r < 8; ++r) {
        gib[(mb + r) * H3 + n] = accI[r] + bi_;
        ghb[(mb + r) * H3 + n] = accH[r] + bh_;
      }
    }
    wg_sync();
    // ---- S7: GRU update + store output ------------------------------------
    for (int i = tid; i < B_ * H_; i += 1024) {
      int b = i >> 7, j = i & 127;
      float r_  = sigmoidf_(gib[b * H3 + j]       + ghb[b * H3 + j]);
      float zz  = sigmoidf_(gib[b * H3 + 128 + j] + ghb[b * H3 + 128 + j]);
      float nn  = tanhf(gib[b * H3 + 256 + j] + r_ * ghb[b * H3 + 256 + j]);
      float hv  = hbuf[i];
      float hn  = (1.0f - zz) * nn + zz * hv;
      hbuf[i]  = hn;
      s_hbf[i] = f2bf(hn);
      if (dir == 0) {
        Ar.out[((size_t)s * B_ + b) * (2 * H_) + j] = Ar.cmask[b * T_ + s] ? hn : 0.0f;
      } else {
        int L = s_len[b];
        int p = (s < L) ? (L - 1 - s) : s;                  // inverse masked_flip
        Ar.out[((size_t)p * B_ + b) * (2 * H_) + H_ + j] = (s < L) ? hn : 0.0f;
      }
    }
    wg_sync();
  }
}

// ---------------------------------------------------------------------------
extern "C" void kernel_launch(void* const* d_in, const int* in_sizes, int n_in,
                              void* d_out, int out_size, void* d_ws, size_t ws_size,
                              hipStream_t stream) {
  enum { WQ_W = 0, WQ_B, WP_W, WP_B, WR_W, WR_B, WG_W, WG_B,
         GL_W, GL_B, LN_G, LN_B, W_IH, B_IH, W_HH, B_HH };
  const float* cv    = (const float*)d_in[0];
  const int*   cmask = (const int*)d_in[1];
  const float* qv    = (const float*)d_in[2];
  const int*   qmask = (const int*)d_in[3];
  const float* P[2][16];
  for (int d = 0; d < 2; ++d)
    for (int k = 0; k < 16; ++k) P[d][k] = (const float*)d_in[4 + d * 16 + k];

  char* ws = (char*)d_ws;
  size_t off = 0;
  auto take = [&](size_t bytes) -> char* {
    char* p = ws + off;
    off += (bytes + 255) & ~(size_t)255;
    return p;
  };

  StepArgs A;
  A.cv = cv; A.cmask = cmask; A.qmask = qmask; A.out = (float*)d_out;
  float* wh[2]; float* wp[2];
  uint16_t* wrwb[2]; uint16_t* glwb[2]; uint16_t* wihb[2]; uint16_t* whhb[2];
  uint16_t* wqwb[2]; uint16_t* wpwb[2];
  for (int d = 0; d < 2; ++d) {
    wh[d]   = (float*)take((size_t)Q_ * B_ * H_ * 4);
    wp[d]   = (float*)take((size_t)T_ * B_ * H_ * 4);
    wrwb[d] = (uint16_t*)take((size_t)H_ * H_ * 2);
    glwb[d] = (uint16_t*)take((size_t)D2 * D2 * 2);
    wihb[d] = (uint16_t*)take((size_t)H3 * D2 * 2);
    whhb[d] = (uint16_t*)take((size_t)H3 * H_ * 2);
    wqwb[d] = (uint16_t*)take((size_t)H_ * D_ * 2);
    wpwb[d] = (uint16_t*)take((size_t)H_ * D_ * 2);
    A.h[d]   = (float*)take((size_t)B_ * H_ * 4);
    A.wrh[d] = (float*)take((size_t)B_ * H_ * 4);
    A.z[d]   = (float*)take((size_t)B_ * D2 * 4);
    A.zg[d]  = (float*)take((size_t)B_ * D2 * 4);
    A.gi[d]  = (float*)take((size_t)B_ * H3 * 4);
    A.gh[d]  = (float*)take((size_t)B_ * H3 * 4);
    A.zbf[d] = (uint16_t*)take((size_t)B_ * D2 * 2);
  }
  uint16_t* qvb = (uint16_t*)take((size_t)Q_ * B_ * D_ * 2);
  uint16_t* cvb = (uint16_t*)take((size_t)T_ * B_ * D_ * 2);
  A.qv_bf = qvb;
  for (int d = 0; d < 2; ++d) {
    A.wh[d] = wh[d];   A.wp[d] = wp[d];
    A.wrw[d] = wrwb[d]; A.glw[d] = glwb[d]; A.wih[d] = wihb[d]; A.whh[d] = whhb[d];
    A.wr_b[d] = P[d][WR_B]; A.wg_w[d] = P[d][WG_W]; A.wg_b[d] = P[d][WG_B];
    A.gl_b[d] = P[d][GL_B]; A.ln_g[d] = P[d][LN_G]; A.ln_b[d] = P[d][LN_B];
    A.b_ih[d] = P[d][B_IH]; A.b_hh[d] = P[d][B_HH];
  }

  // ---- Phase A: one-time bf16 conversions, then pure-bf16 WMMA GEMMs ------
  auto cvt = [&](const float* s, uint16_t* dst, int n) {
    k_cvt_bf16<<<(n + 255) / 256, 256, 0, stream>>>(s, dst, n);
  };
  cvt(qv, qvb, Q_ * B_ * D_);
  cvt(cv, cvb, T_ * B_ * D_);
  for (int d = 0; d < 2; ++d) {
    cvt(P[d][WR_W], wrwb[d], H_ * H_);
    cvt(P[d][GL_W], glwb[d], D2 * D2);
    cvt(P[d][W_IH], wihb[d], H3 * D2);
    cvt(P[d][W_HH], whhb[d], H3 * H_);
    cvt(P[d][WQ_W], wqwb[d], H_ * D_);
    cvt(P[d][WP_W], wpwb[d], H_ * D_);
  }
  for (int d = 0; d < 2; ++d) {
    // wh = qv.(QB,D) @ wq_w^T + wq_b : M=4096, N=128, K=256 -> 2048 tiles
    k_gemm_bf16<<<(4096 / 16) * (128 / 16) / 8, 256, 0, stream>>>(
        qvb, wqwb[d], P[d][WQ_B], wh[d], Q_ * B_, H_, D_);
    // wp = cv.(TB,D) @ wp_w^T + wp_b : M=24576, N=128, K=256 -> 12288 tiles
    k_gemm_bf16<<<(24576 / 16) * (128 / 16) / 8, 256, 0, stream>>>(
        cvb, wpwb[d], P[d][WP_B], wp[d], T_ * B_, H_, D_);
  }

  // ---- Phase B: persistent recurrent scan, one workgroup per direction ----
  k_matchrnn<<<2, 1024, 0, stream>>>(A);
}
